// FujiGatedDeltaNet_26259430048066
// MI455X (gfx1250) — compile-verified, matrix operand order checked by reading, not run on hardware
//
#include <hip/hip_runtime.h>
#include <hip/hip_bf16.h>
#include <math.h>

// ---------------- problem constants ----------------
constexpr int SEQ    = 2048;
constexpr int BATCH  = 2;
constexpr int BS     = SEQ * BATCH;          // 4096 rows
constexpr int HIDDEN = 2048;
constexpr int HK = 16, HV = 32, DK = 128, DV = 128;
constexpr int RATIO  = HV / HK;              // 2
constexpr int KEY_DIM   = HK * DK;           // 2048
constexpr int VALUE_DIM = HV * DV;           // 4096
constexpr int CONV_DIM  = 2 * KEY_DIM + VALUE_DIM;   // 8192
constexpr int QKVZ_N    = 2 * KEY_DIM + 2 * VALUE_DIM; // 12288
constexpr int PERHEAD   = QKVZ_N / HK;       // 768 = [q128|k128|v256|z256]
constexpr int KCONV  = 4;
constexpr int CHUNKC = 64;
constexpr int NCHUNK = SEQ / CHUNKC;         // 32
constexpr float EPSF = 1e-6f;

typedef __attribute__((ext_vector_type(16))) __bf16 bf16x16;
typedef __attribute__((ext_vector_type(8)))  __bf16 bf16x8;
typedef __attribute__((ext_vector_type(8)))  float  f32x8;

// ---------------- WMMA helpers ----------------
__device__ __forceinline__ f32x8 wmma_bf16(bf16x16 a, bf16x16 b, f32x8 c) {
  return __builtin_amdgcn_wmma_f32_16x16x32_bf16(false, a, false, b, (short)0, c,
                                                 false, false);
}

// Load a 16-lane bf16 fragment from a K-major matrix (row stride = ld elements).
// A-operand: row = M index.  B-operand: row = N index (matrix stored as B^T, [n][k]).
// Per ISA 16-bit layout: lanes 0-15 hold K = kb+0..7 and kb+16..23,
// lanes 16-31 hold K = kb+8..15 and kb+24..31 (two contiguous 16B runs).
__device__ __forceinline__ bf16x16 ldfrag(const __bf16* base, int ld, int row, int kbase) {
  const int lane = threadIdx.x & 31;
  const __bf16* p = base + (size_t)(row + (lane & 15)) * ld + kbase + ((lane & 16) ? 8 : 0);
  bf16x8 lo = *(const bf16x8*)(p);
  bf16x8 hi = *(const bf16x8*)(p + 16);
  return __builtin_shufflevector(lo, hi, 0, 1, 2, 3, 4, 5, 6, 7, 8, 9, 10, 11, 12, 13, 14, 15);
}

// ---------------- elementwise prep kernels ----------------
// (S,B,H) f32 -> (B*S,H) bf16
__global__ void transpose_cast_h(const float* __restrict__ h, __bf16* __restrict__ out) {
  size_t idx = (size_t)blockIdx.x * blockDim.x + threadIdx.x;
  if (idx >= (size_t)SEQ * BATCH * HIDDEN) return;
  int c = idx % HIDDEN;
  size_t sb = idx / HIDDEN;
  int b = sb % BATCH;
  int s = sb / BATCH;
  out[((size_t)(b * SEQ + s)) * HIDDEN + c] = (__bf16)h[idx];
}

__global__ void cast_bf16(const float* __restrict__ in, __bf16* __restrict__ out, size_t n) {
  size_t idx = (size_t)blockIdx.x * blockDim.x + threadIdx.x;
  if (idx < n) out[idx] = (__bf16)in[idx];
}

// ---------------- generic WMMA GEMM:  C[M,N] = A[M,K] * Bt[N,K]^T ----------------
// block = 256 threads = 8 waves (4 M-waves x 2 N-waves); block tile 64M x 128N;
// each wave computes 16M x 64N (4 accumulators).
__global__ __launch_bounds__(256) void gemm_bf16_wmma(
    const __bf16* __restrict__ A, const __bf16* __restrict__ Bt, float* __restrict__ C,
    int M, int N, int K, int out_mode /*0: row-major; 1: (s,b,n) swap*/) {
  const int lane = threadIdx.x & 31;
  const int w = threadIdx.x >> 5;
  const int m0 = blockIdx.y * 64 + (w >> 1) * 16;
  const int n0 = blockIdx.x * 128 + (w & 1) * 64;
  f32x8 acc0 = {}, acc1 = {}, acc2 = {}, acc3 = {};
  for (int k = 0; k < K; k += 32) {
    bf16x16 af = ldfrag(A, K, m0, k);
    acc0 = wmma_bf16(af, ldfrag(Bt, K, n0 + 0,  k), acc0);
    acc1 = wmma_bf16(af, ldfrag(Bt, K, n0 + 16, k), acc1);
    acc2 = wmma_bf16(af, ldfrag(Bt, K, n0 + 32, k), acc2);
    acc3 = wmma_bf16(af, ldfrag(Bt, K, n0 + 48, k), acc3);
  }
  const int mb = m0 + ((lane & 16) ? 8 : 0);
  f32x8 accs[4] = {acc0, acc1, acc2, acc3};
#pragma unroll
  for (int t = 0; t < 4; ++t) {
    int n = n0 + t * 16 + (lane & 15);
#pragma unroll
    for (int r = 0; r < 8; ++r) {
      int m = mb + r;
      float v = accs[t][r];
      if (out_mode == 0) {
        C[(size_t)m * N + n] = v;
      } else {  // row m = b*SEQ + s  ->  out[(s*BATCH + b)*N + n]
        int b = m / SEQ, s = m % SEQ;
        C[((size_t)s * BATCH + b) * N + n] = v;
      }
    }
  }
}

// ---------------- ba projection + beta / g ----------------
__global__ void ba_beta_g(const __bf16* __restrict__ hbf, const float* __restrict__ Wba,
                          const float* __restrict__ dt_bias, const float* __restrict__ A_log,
                          float* __restrict__ betab, float* __restrict__ gb) {
  const int m = blockIdx.x;        // 0..BS-1
  const int t = threadIdx.x;       // 0..63  (column of W_ba)
  const __bf16* hr = hbf + (size_t)m * HIDDEN;
  const float* wr = Wba + (size_t)t * HIDDEN;
  float acc = 0.f;
#pragma unroll 8
  for (int k = 0; k < HIDDEN; ++k) acc += (float)hr[k] * wr[k];
  const int b = m / SEQ, s = m % SEQ;
  const int hk = t >> 2, r = t & 3;  // ba layout (HK, 2*RATIO)
  if (r < 2) {
    int hv = hk * 2 + r;
    betab[((size_t)(b * HV + hv)) * SEQ + s] = 1.f / (1.f + __expf(-acc));
  } else {
    int hv = hk * 2 + (r - 2);
    float x = acc + dt_bias[hv];
    float sp = (x > 20.f) ? x : log1pf(__expf(x));
    gb[((size_t)(b * HV + hv)) * SEQ + s] = -__expf(A_log[hv]) * sp;
  }
}

// ---------------- depthwise conv (K=4) + SiLU, scatter to head-major layouts ----------------
__global__ void conv_silu(const float* __restrict__ qkvz, const float* __restrict__ convw,
                          float* __restrict__ qc, float* __restrict__ kc, float* __restrict__ vc) {
  size_t idx = (size_t)blockIdx.x * blockDim.x + threadIdx.x;
  if (idx >= (size_t)BATCH * SEQ * CONV_DIM) return;
  const int c = idx % CONV_DIM;
  size_t bs = idx / CONV_DIM;
  const int s = bs % SEQ;
  const int b = bs / SEQ;
  // map mixed channel c -> column in the per-head-interleaved qkvz row
  int col;
  if (c < KEY_DIM)           { int h = c >> 7;  col = h * PERHEAD + (c & 127); }
  else if (c < 2 * KEY_DIM)  { int cc = c - KEY_DIM; int h = cc >> 7; col = h * PERHEAD + 128 + (cc & 127); }
  else                       { int cc = c - 2 * KEY_DIM; int h = cc >> 8; col = h * PERHEAD + 256 + (cc & 255); }
  float acc = 0.f;
#pragma unroll
  for (int i = 0; i < KCONV; ++i) {
    int t = s - (KCONV - 1) + i;
    if (t >= 0) acc += qkvz[((size_t)(b * SEQ + t)) * QKVZ_N + col] * convw[c * KCONV + i];
  }
  float y = acc / (1.f + __expf(-acc));  // silu
  if (c < KEY_DIM) {
    int h = c >> 7;
    qc[(((size_t)(b * HK + h)) * SEQ + s) * DK + (c & 127)] = y;
  } else if (c < 2 * KEY_DIM) {
    int cc = c - KEY_DIM; int h = cc >> 7;
    kc[(((size_t)(b * HK + h)) * SEQ + s) * DK + (cc & 127)] = y;
  } else {
    int cc = c - 2 * KEY_DIM; int hv = cc >> 7;
    vc[(((size_t)(b * HV + hv)) * SEQ + s) * DV + (cc & 127)] = y;
  }
}

// ---------------- chunked gated delta-rule core ----------------
// One workgroup (8 wave32) per (b,hv).  Persistent 128x128 f32 state + bf16 mirror in LDS.
// LDS byte offsets (total 295424 B, fits CDNA5's 320KB/WGP):
constexpr int O_SF32 = 0;                       // f32 state (dv,dk)       64KB
constexpr int O_SBF  = O_SF32 + 128 * 128 * 4;  // bf16 state mirror       32KB
constexpr int O_QN   = O_SBF + 128 * 128 * 2;   // qn  (c,dk) bf16         16KB
constexpr int O_QG   = O_QN + 64 * 128 * 2;     // qn*exp(gc)              16KB
constexpr int O_KN   = O_QG + 64 * 128 * 2;     // kn  (c,dk)              16KB
constexpr int O_KB   = O_KN + 64 * 128 * 2;     // kn*beta                 16KB
constexpr int O_KBGT = O_KB + 64 * 128 * 2;     // (kb*exp(gc))^T (dk,c)   16KB
constexpr int O_KDT  = O_KBGT + 128 * 64 * 2;   // (kn*exp(gl-gc))^T       16KB
constexpr int O_VBT  = O_KDT + 128 * 64 * 2;    // (v*beta)^T (dv,c)       16KB
constexpr int O_VNT  = O_VBT + 128 * 64 * 2;    // v_new^T (dv,c)          16KB
constexpr int O_KCUM = O_VNT + 128 * 64 * 2;    // k_cum (c,dk)            16KB
constexpr int O_AF   = O_KCUM + 64 * 128 * 2;   // A f32 (64x64)           16KB
constexpr int O_TF   = O_AF + 64 * 64 * 4;      // T f32                   16KB
constexpr int O_TBF  = O_TF + 64 * 64 * 4;      // T bf16                   8KB
constexpr int O_AIBF = O_TBF + 64 * 64 * 2;     // A_incl bf16              8KB
constexpr int O_GC   = O_AIBF + 64 * 64 * 2;    // gc[64] f32
constexpr int SMEM_BYTES = O_GC + 64 * 4 + 256;

__global__ __launch_bounds__(256) void gdn_core(
    const float* __restrict__ qc, const float* __restrict__ kc, const float* __restrict__ vc,
    const float* __restrict__ gb, const float* __restrict__ betab, float* __restrict__ core) {
  extern __shared__ char smem[];
  float*  s_f32 = (float*)(smem + O_SF32);
  __bf16* s_bf  = (__bf16*)(smem + O_SBF);
  __bf16* qn    = (__bf16*)(smem + O_QN);
  __bf16* qg    = (__bf16*)(smem + O_QG);
  __bf16* kn    = (__bf16*)(smem + O_KN);
  __bf16* kb    = (__bf16*)(smem + O_KB);
  __bf16* kbgT  = (__bf16*)(smem + O_KBGT);
  __bf16* kdT   = (__bf16*)(smem + O_KDT);
  __bf16* vbT   = (__bf16*)(smem + O_VBT);
  __bf16* vnT   = (__bf16*)(smem + O_VNT);
  __bf16* kcum  = (__bf16*)(smem + O_KCUM);
  float*  a_f   = (float*)(smem + O_AF);
  float*  t_f   = (float*)(smem + O_TF);
  __bf16* t_bf  = (__bf16*)(smem + O_TBF);
  __bf16* ai_bf = (__bf16*)(smem + O_AIBF);
  float*  gcs   = (float*)(smem + O_GC);

  const int bh = blockIdx.x;                   // b*HV + hv
  const int b = bh / HV, hv = bh % HV, hk = hv >> 1;
  const int tid = threadIdx.x, lane = tid & 31, w = tid >> 5;
  const int laneHi = (lane & 16) ? 8 : 0;

  const float* qrow = qc + ((size_t)(b * HK + hk)) * SEQ * DK;
  const float* krow = kc + ((size_t)(b * HK + hk)) * SEQ * DK;
  const float* vrow = vc + ((size_t)(b * HV + hv)) * SEQ * DV;
  const float* gp   = gb + ((size_t)(b * HV + hv)) * SEQ;
  const float* bp   = betab + ((size_t)(b * HV + hv)) * SEQ;

  for (int i = tid; i < 128 * 128; i += 256) { s_f32[i] = 0.f; s_bf[i] = (__bf16)0.f; }
  __syncthreads();

  for (int nc = 0; nc < NCHUNK; ++nc) {
    const int s0 = nc * CHUNKC;
    if (tid == 0) {
      float a = 0.f;
      for (int i = 0; i < 64; ++i) { a += gp[s0 + i]; gcs[i] = a; }
    }
    __syncthreads();
    const float glast = gcs[63];

    // ---- per-row prep: l2norm, beta/decay scalings, transposed staging ----
    for (int r = w; r < 64; r += 8) {
      const float eg = __expf(gcs[r]);
      const float ed = __expf(glast - gcs[r]);
      const float beta_r = bp[s0 + r];
      const int c0 = lane * 4;
      // k
      float4 kv = ((const float4*)(krow + (size_t)(s0 + r) * DK))[lane];
      float ss = kv.x * kv.x + kv.y * kv.y + kv.z * kv.z + kv.w * kv.w;
      for (int m = 16; m >= 1; m >>= 1) ss += __shfl_xor(ss, m, 32);
      float rn = rsqrtf(ss + EPSF);
      float kvv[4] = {kv.x * rn, kv.y * rn, kv.z * rn, kv.w * rn};
#pragma unroll
      for (int j = 0; j < 4; ++j) {
        float knv = kvv[j], kbv = knv * beta_r;
        kn[r * DK + c0 + j] = (__bf16)knv;
        kb[r * DK + c0 + j] = (__bf16)kbv;
        kbgT[(c0 + j) * 64 + r] = (__bf16)(kbv * eg);
        kdT[(c0 + j) * 64 + r] = (__bf16)(knv * ed);
      }
      // q (l2norm + dk^-0.5)
      float4 qv = ((const float4*)(qrow + (size_t)(s0 + r) * DK))[lane];
      float qs = qv.x * qv.x + qv.y * qv.y + qv.z * qv.z + qv.w * qv.w;
      for (int m = 16; m >= 1; m >>= 1) qs += __shfl_xor(qs, m, 32);
      float qrn = rsqrtf(qs + EPSF) * 0.08838834764831845f;
      float qvv[4] = {qv.x * qrn, qv.y * qrn, qv.z * qrn, qv.w * qrn};
#pragma unroll
      for (int j = 0; j < 4; ++j) {
        qn[r * DK + c0 + j] = (__bf16)qvv[j];
        qg[r * DK + c0 + j] = (__bf16)(qvv[j] * eg);
      }
      // v*beta transposed
      float4 vv = ((const float4*)(vrow + (size_t)(s0 + r) * DV))[lane];
      float vvv[4] = {vv.x * beta_r, vv.y * beta_r, vv.z * beta_r, vv.w * beta_r};
#pragma unroll
      for (int j = 0; j < 4; ++j) vbT[(c0 + j) * 64 + r] = (__bf16)vvv[j];
    }
    __syncthreads();

    // ---- A = -(kb @ kn^T) * decay (strict lower) ----
    for (int t = w * 2; t < w * 2 + 2; ++t) {
      const int tm = t >> 2, tn = t & 3;
      f32x8 acc = {};
      for (int kk = 0; kk < 128; kk += 32)
        acc = wmma_bf16(ldfrag(kb, 128, tm * 16, kk), ldfrag(kn, 128, tn * 16, kk), acc);
      const int nl = tn * 16 + (lane & 15), mb = tm * 16 + laneHi;
#pragma unroll
      for (int r = 0; r < 8; ++r) {
        int m = mb + r;
        a_f[m * 64 + nl] = (nl < m) ? -acc[r] * __expf(gcs[m] - gcs[nl]) : 0.f;
      }
    }
    __syncthreads();

    // ---- T = (I - A)^{-1}: column-parallel forward substitution ----
    if (tid < 64) {
      const int j = tid;
      for (int i = 0; i < j; ++i) { t_f[i * 64 + j] = 0.f; t_bf[i * 64 + j] = (__bf16)0.f; }
      t_f[j * 64 + j] = 1.f; t_bf[j * 64 + j] = (__bf16)1.f;
      for (int i = j + 1; i < 64; ++i) {
        float acc = a_f[i * 64 + j];
        for (int l = j + 1; l < i; ++l) acc += a_f[i * 64 + l] * t_f[l * 64 + j];
        t_f[i * 64 + j] = acc;
        t_bf[i * 64 + j] = (__bf16)acc;
      }
    }
    __syncthreads();

    // ---- k_cum = T @ (kb * exp(gc)) ----
    for (int t = w * 4; t < w * 4 + 4; ++t) {
      const int tm = t >> 3, tn = t & 7;
      f32x8 acc = {};
      for (int kk = 0; kk < 64; kk += 32)
        acc = wmma_bf16(ldfrag(t_bf, 64, tm * 16, kk), ldfrag(kbgT, 64, tn * 16, kk), acc);
      const int nl = tn * 16 + (lane & 15), mb = tm * 16 + laneHi;
#pragma unroll
      for (int r = 0; r < 8; ++r) kcum[(mb + r) * 128 + nl] = (__bf16)acc[r];
    }
    __syncthreads();

    // ---- v_new = T @ vb - k_cum @ state   (stored transposed) ----
    for (int t = w * 4; t < w * 4 + 4; ++t) {
      const int tm = t >> 3, tn = t & 7;
      f32x8 a1 = {}, a2 = {};
      for (int kk = 0; kk < 64; kk += 32)
        a1 = wmma_bf16(ldfrag(t_bf, 64, tm * 16, kk), ldfrag(vbT, 64, tn * 16, kk), a1);
      for (int kk = 0; kk < 128; kk += 32)
        a2 = wmma_bf16(ldfrag(kcum, 128, tm * 16, kk), ldfrag(s_bf, 128, tn * 16, kk), a2);
      const int nl = tn * 16 + (lane & 15), mb = tm * 16 + laneHi;
#pragma unroll
      for (int r = 0; r < 8; ++r) vnT[nl * 64 + mb + r] = (__bf16)(a1[r] - a2[r]);
    }
    __syncthreads();

    // ---- A_incl = (q @ k^T) * decay (inclusive) ----
    for (int t = w * 2; t < w * 2 + 2; ++t) {
      const int tm = t >> 2, tn = t & 3;
      f32x8 acc = {};
      for (int kk = 0; kk < 128; kk += 32)
        acc = wmma_bf16(ldfrag(qn, 128, tm * 16, kk), ldfrag(kn, 128, tn * 16, kk), acc);
      const int nl = tn * 16 + (lane & 15), mb = tm * 16 + laneHi;
#pragma unroll
      for (int r = 0; r < 8; ++r) {
        int m = mb + r;
        ai_bf[m * 64 + nl] = (__bf16)((nl <= m) ? acc[r] * __expf(gcs[m] - gcs[nl]) : 0.f);
      }
    }
    __syncthreads();

    // ---- out = (q*exp(gc)) @ state + A_incl @ v_new ----
    for (int t = w * 4; t < w * 4 + 4; ++t) {
      const int tm = t >> 3, tn = t & 7;
      f32x8 acc = {};
      for (int kk = 0; kk < 128; kk += 32)
        acc = wmma_bf16(ldfrag(qg, 128, tm * 16, kk), ldfrag(s_bf, 128, tn * 16, kk), acc);
      for (int kk = 0; kk < 64; kk += 32)
        acc = wmma_bf16(ldfrag(ai_bf, 64, tm * 16, kk), ldfrag(vnT, 64, tn * 16, kk), acc);
      const int nl = tn * 16 + (lane & 15), mb = tm * 16 + laneHi;
#pragma unroll
      for (int r = 0; r < 8; ++r) {
        int m = mb + r;
        core[(((size_t)b * SEQ + s0 + m) * HV + hv) * DV + nl] = acc[r];
      }
    }
    __syncthreads();

    // ---- state = exp(g_last)*state + v_new^T @ k_dec  (stored (dv,dk)) ----
    const float egl = __expf(glast);
    for (int t = w * 8; t < w * 8 + 8; ++t) {
      const int tm = t >> 3, tn = t & 7;     // tm over dv, tn over dk
      const int nl = tn * 16 + (lane & 15), mb = tm * 16 + laneHi;
      f32x8 acc;
#pragma unroll
      for (int r = 0; r < 8; ++r) acc[r] = egl * s_f32[(mb + r) * 128 + nl];
      for (int kk = 0; kk < 64; kk += 32)
        acc = wmma_bf16(ldfrag(vnT, 64, tm * 16, kk), ldfrag(kdT, 64, tn * 16, kk), acc);
#pragma unroll
      for (int r = 0; r < 8; ++r) {
        s_f32[(mb + r) * 128 + nl] = acc[r];
        s_bf[(mb + r) * 128 + nl] = (__bf16)acc[r];
      }
    }
    __syncthreads();
  }
}

// ---------------- RMSNorm + SiLU(z) gate -> bf16 rows for output GEMM ----------------
__global__ void norm_gate(const float* __restrict__ core, const float* __restrict__ qkvz,
                          const float* __restrict__ norm_w, __bf16* __restrict__ G) {
  const int row = blockIdx.x;        // (b*SEQ+s)*HV + hv
  const int d = threadIdx.x;         // 0..127
  const int hv = row % HV;
  const size_t bs = row / HV;        // b*SEQ + s
  float x = core[(size_t)row * DV + d];
  __shared__ float red[4];
  float ss = x * x;
  for (int m = 16; m >= 1; m >>= 1) ss += __shfl_xor(ss, m, 32);
  if ((d & 31) == 0) red[d >> 5] = ss;
  __syncthreads();
  float tot = red[0] + red[1] + red[2] + red[3];
  float xr = x * rsqrtf(tot * (1.f / 128.f) + EPSF) * norm_w[d];
  const int hk = hv >> 1, part = hv & 1;
  float z = qkvz[bs * QKVZ_N + hk * PERHEAD + 512 + part * 128 + d];
  float zg = z / (1.f + __expf(-z));
  G[bs * VALUE_DIM + (size_t)hv * DV + d] = (__bf16)(xr * zg);
}

// ---------------- host launch ----------------
extern "C" void kernel_launch(void* const* d_in, const int* in_sizes, int n_in,
                              void* d_out, int out_size, void* d_ws, size_t ws_size,
                              hipStream_t stream) {
  (void)in_sizes; (void)n_in; (void)out_size; (void)ws_size;
  const float* hid    = (const float*)d_in[0];  // (S,B,HIDDEN)
  const float* Wqkvz  = (const float*)d_in[1];  // (12288,2048)
  const float* Wba    = (const float*)d_in[2];  // (64,2048)
  const float* convw  = (const float*)d_in[3];  // (8192,4)
  const float* dtb    = (const float*)d_in[4];  // (32,)
  const float* A_log  = (const float*)d_in[5];  // (32,)
  const float* norm_w = (const float*)d_in[6];  // (128,)
  const float* Wout   = (const float*)d_in[7];  // (2048,4096)
  float* out = (float*)d_out;                   // (S,B,2048)

  // workspace carve-up (~497 MB)
  char* ws = (char*)d_ws;
  __bf16* hbf     = (__bf16*)(ws + 0);                     // 16 MB
  __bf16* wqkvz_b = (__bf16*)(ws + 16777216ull);           // 48 MB
  __bf16* wout_b  = (__bf16*)(ws + 67108864ull);           // 16 MB
  float*  qkvz    = (float*) (ws + 83886080ull);           // 192 MB
  float*  qc      = (float*) (ws + 285212672ull);          // 32 MB
  float*  kc      = (float*) (ws + 318767104ull);          // 32 MB
  float*  vc      = (float*) (ws + 352321536ull);          // 64 MB
  float*  betab   = (float*) (ws + 419430400ull);          // 0.5 MB
  float*  gb      = (float*) (ws + 419954688ull);          // 0.5 MB
  float*  corebuf = (float*) (ws + 420478976ull);          // 64 MB
  __bf16* gated   = (__bf16*)(ws + 487587840ull);          // 32 MB

  // 0) casts / transpose
  {
    size_t n = (size_t)SEQ * BATCH * HIDDEN;
    transpose_cast_h<<<(unsigned)((n + 255) / 256), 256, 0, stream>>>(hid, hbf);
  }
  {
    size_t n = (size_t)QKVZ_N * HIDDEN;
    cast_bf16<<<(unsigned)((n + 255) / 256), 256, 0, stream>>>(Wqkvz, wqkvz_b, n);
  }
  {
    size_t n = (size_t)HIDDEN * VALUE_DIM;
    cast_bf16<<<(unsigned)((n + 255) / 256), 256, 0, stream>>>(Wout, wout_b, n);
  }

  // 1) qkvz projection GEMM: (4096 x 2048) x (2048 x 12288)
  gemm_bf16_wmma<<<dim3(QKVZ_N / 128, BS / 64), 256, 0, stream>>>(
      hbf, wqkvz_b, qkvz, BS, QKVZ_N, HIDDEN, 0);

  // 2) ba projection + beta / g
  ba_beta_g<<<BS, 64, 0, stream>>>(hbf, Wba, dtb, A_log, betab, gb);

  // 3) depthwise conv + silu
  {
    size_t n = (size_t)BATCH * SEQ * CONV_DIM;
    conv_silu<<<(unsigned)((n + 255) / 256), 256, 0, stream>>>(qkvz, convw, qc, kc, vc);
  }

  // 4) gated delta-rule core: one workgroup per (b, head), 288 KB LDS
  (void)hipFuncSetAttribute((const void*)gdn_core,
                            hipFuncAttributeMaxDynamicSharedMemorySize, SMEM_BYTES);
  gdn_core<<<BATCH * HV, 256, SMEM_BYTES, stream>>>(qc, kc, vc, gb, betab, corebuf);

  // 5) RMSNorm + silu(z) gate
  norm_gate<<<BATCH * SEQ * HV, 128, 0, stream>>>(corebuf, qkvz, norm_w, gated);

  // 6) output projection GEMM: (4096 x 4096) x (4096 x 2048), writes (S,B,H)
  gemm_bf16_wmma<<<dim3(HIDDEN / 128, BS / 64), 256, 0, stream>>>(
      gated, wout_b, out, BS, HIDDEN, VALUE_DIM, 1);
}
